// HierForecastNet_35270271434906
// MI455X (gfx1250) — matrix operand order repaired and checked by reading, not compiled, not run on hardware
//
#include <hip/hip_runtime.h>
#include <hip/hip_bf16.h>
#include <utility>

// ---------------------------------------------------------------------------
// HierForecastNet fused CDNA5 implementation.
// - All GEMMs: V_WMMA_F32_16X16X32_F16 (f16 A/B, f32 accum, C init inline 0,
//   bias folded into epilogue).
// - M_r=2 batch blocking (32 samples/wave): each 1KB weight B-tile load is
//   amortized over 2-4 WMMAs (weight operand traffic is the binding resource).
// - Depth-2 software-pipelined B-tile loads in the LSTM kernels, expressed
//   via compile-time static_for so the MAC phase is guaranteed straight-line
//   code (no rolled loops / dynamic accumulator indexing).
// - Scalar phases (softmax / heads / attention) use v8h / float4 vector
//   LDS+global accesses and run one row per lane on all 32 lanes.
// ---------------------------------------------------------------------------

typedef __attribute__((ext_vector_type(16))) _Float16 v16h;
typedef __attribute__((ext_vector_type(8)))  _Float16 v8h;
typedef __attribute__((ext_vector_type(8)))  float    v8f;
typedef __attribute__((ext_vector_type(4)))  float    v4f;
typedef _Float16 h16;

#define DEVI __device__ __forceinline__

// Compile-time loop: guarantees full expansion (frontend), unlike #pragma.
template <typename F, int... Is>
DEVI void sfor_impl(F&& f, std::integer_sequence<int, Is...>) {
  (f(std::integral_constant<int, Is>{}), ...);
}
template <int N, typename F>
DEVI void sfor(F&& f) { sfor_impl(f, std::make_integer_sequence<int, N>{}); }

static constexpr int B_  = 32768;
static constexpr int T_  = 14;
static constexpr int H_  = 128;
static constexpr int ST_ = 30;

// ---- workspace byte offsets (all 256B aligned) ----------------------------
static constexpr size_t OFF_SEQ   = 0;                                   // B*14*128 f16 (xw, then hs in place)
static constexpr size_t SZ_SEQ    = (size_t)B_*T_*H_*2;
static constexpr size_t OFF_TOK   = OFF_SEQ + SZ_SEQ;                    // B*128 f16 monthly token
static constexpr size_t SZ_TOK    = (size_t)B_*H_*2;
static constexpr size_t SZ_SQ     = 128*128*2;                           // 128x128 swizzled f16
static constexpr size_t SZ_BIG    = 512*128*2;                           // 512x128 swizzled f16
static constexpr size_t OFF_FA    = OFF_TOK   + SZ_TOK;
static constexpr size_t OFF_EWIH  = OFF_FA    + SZ_SQ;
static constexpr size_t OFF_EWHH  = OFF_EWIH  + SZ_BIG;
static constexpr size_t OFF_WK    = OFF_EWHH  + SZ_BIG;
static constexpr size_t OFF_WV    = OFF_WK    + SZ_SQ;
static constexpr size_t OFF_WOUT  = OFF_WV    + SZ_SQ;
static constexpr size_t OFF_DWHH  = OFF_WOUT  + SZ_SQ;
static constexpr size_t OFF_DWSUM = OFF_DWHH  + SZ_BIG;
static constexpr size_t OFF_Q     = OFF_DWSUM + SZ_BIG;                  // 128 f32
static constexpr size_t OFF_EBSUM = OFF_Q     + 128*4;                   // 512 f32
static constexpr size_t OFF_DBSUM = OFF_EBSUM + 512*4;                   // 512 f32

// ---- WMMA helpers ---------------------------------------------------------
DEVI v8f wmma_ff(v16h a, v16h b, v8f c) {
  return __builtin_amdgcn_wmma_f32_16x16x32_f16(false, a, false, b, (short)0, c, false, false);
}
DEVI v16h zero_a() { v16h a;
#pragma unroll
  for (int i = 0; i < 16; ++i) a[i] = (h16)0.f;
  return a;
}
DEVI v8f zero_c() { v8f c;
#pragma unroll
  for (int i = 0; i < 8; ++i) c[i] = 0.f;
  return c;
}
// A operand (16x32 f16 tile) from a row-major f16 buffer (global or LDS).
DEVI v16h load_a_f16(const h16* base, int rstride, int kb, int lane) {
  const int hsel = (lane >> 4) * 8, r = lane & 15;
  const h16* p = base + (size_t)r * rstride + kb * 32 + hsel;
  v8h lo = *(const v8h*)p;
  v8h hi = *(const v8h*)(p + 16);
  v16h a;
#pragma unroll
  for (int i = 0; i < 8; ++i) { a[i] = lo[i]; a[8 + i] = hi[i]; }
  return a;
}
// Same but from f32 source (x14) with conversion.
DEVI v16h load_a_f32(const float* base, int rstride, int kb, int lane) {
  const int hsel = (lane >> 4) * 8, r = lane & 15;
  const float* p = base + (size_t)r * rstride + kb * 32 + hsel;
  v16h a;
#pragma unroll
  for (int i = 0; i < 8; ++i) { a[i] = (h16)p[i]; a[8 + i] = (h16)p[i + 16]; }
  return a;
}
// B operand from pre-swizzled blocks: block (kb,nt) holds lane-major 16 halfs.
DEVI v16h load_b(const h16* swz, int kb, int nt, int NT, int lane) {
  const h16* p = swz + (((size_t)(kb * NT + nt)) * 32 + lane) * 16;
  v8h lo = *(const v8h*)p;
  v8h hi = *(const v8h*)(p + 8);
  v16h b;
#pragma unroll
  for (int i = 0; i < 8; ++i) { b[i] = lo[i]; b[8 + i] = hi[i]; }
  return b;
}

// ---- scalar math ----------------------------------------------------------
DEVI float sigm_f(float x)  { return 1.f / (1.f + __expf(-x)); }
DEVI float tanh_f(float x)  { x = fminf(fmaxf(x, -15.f), 15.f);
                              float e = __expf(2.f * x); return (e - 1.f) / (e + 1.f); }
DEVI float softp_f(float x) { return x > 20.f ? x : log1pf(__expf(x)); }

// Vectorized two-head dot: row (128 f16 in LDS) . w0 and . w1
DEVI void head_dot(const h16* row, const float* w0, const float* w1,
                   float& acc0, float& acc1) {
#pragma unroll
  for (int c8 = 0; c8 < 16; ++c8) {
    v8h hh = *(const v8h*)(row + c8 * 8);
#pragma unroll
    for (int i = 0; i < 8; ++i) {
      float hv = (float)hh[i];
      acc0 += hv * w0[c8 * 8 + i];
      acc1 += hv * w1[c8 * 8 + i];
    }
  }
}

// ---------------------------------------------------------------------------
// Kernel 0: weight prep (swizzle into WMMA-B lane order, fold dec wih+whh,
// combined biases, q projection).
// ---------------------------------------------------------------------------
DEVI void swz_one(const float* W, const float* W2, int N, int K, h16* dst,
                  int tid, int nth) {
  const int NT = N / 16;
  for (int e = tid; e < N * K; e += nth) {
    int k = e % K, n = e / K;
    float v = W[(size_t)n * K + k];
    if (W2) v += W2[(size_t)n * K + k];
    int kb = k >> 5, kl = k & 31, nt = n >> 4, nl = n & 15;
    int lane = ((kl >> 3) & 1) * 16 + nl;
    int i    = (kl & 7) + ((kl >> 4) << 3);
    dst[(((size_t)(kb * NT + nt)) * 32 + lane) * 16 + i] = (h16)v;
  }
}

__global__ void k_prep(const float* fa_w, const float* enc_wih, const float* enc_whh,
                       const float* pool_win, const float* pool_wout,
                       const float* dec_wih, const float* dec_whh,
                       const float* q_tok, const float* pool_bin,
                       const float* enc_bih, const float* enc_bhh,
                       const float* dec_bih, const float* dec_bhh,
                       h16* fa_s, h16* ewih_s, h16* ewhh_s, h16* wk_s, h16* wv_s,
                       h16* wout_s, h16* dwhh_s, h16* dwsum_s,
                       float* q_s, float* ebsum_s, float* dbsum_s) {
  const int tid = blockIdx.x * blockDim.x + threadIdx.x;
  const int nth = gridDim.x * blockDim.x;
  swz_one(fa_w,            nullptr, 128, 128, fa_s,   tid, nth);
  swz_one(enc_wih,         nullptr, 512, 128, ewih_s, tid, nth);
  swz_one(enc_whh,         nullptr, 512, 128, ewhh_s, tid, nth);
  swz_one(pool_win + 128*128, nullptr, 128, 128, wk_s, tid, nth);
  swz_one(pool_win + 256*128, nullptr, 128, 128, wv_s, tid, nth);
  swz_one(pool_wout,       nullptr, 128, 128, wout_s, tid, nth);
  swz_one(dec_whh,         nullptr, 512, 128, dwhh_s, tid, nth);
  swz_one(dec_wih,         dec_whh, 512, 128, dwsum_s, tid, nth); // x_t==h_t fold
  for (int j = tid; j < 128; j += nth) {
    float a = pool_bin[j];
    for (int k = 0; k < 128; ++k) a += q_tok[k] * pool_win[(size_t)j * 128 + k];
    q_s[j] = a;
  }
  for (int n = tid; n < 512; n += nth) {
    ebsum_s[n] = enc_bih[n] + enc_bhh[n];
    dbsum_s[n] = dec_bih[n] + dec_bhh[n];
  }
}

// ---------------------------------------------------------------------------
// Kernel 1: feature attention. xw = softmax(x14@fa_w.T + fa_b) * x14 -> seq
// 4 waves/WG, 32 token-rows per wave (M_r=2).
// ---------------------------------------------------------------------------
__global__ void __launch_bounds__(128, 1)
k_feat(const float* __restrict__ x14, const h16* __restrict__ fa_s,
       const float* __restrict__ fa_b, h16* __restrict__ seq) {
  __shared__ float S1[4][32 * 128];
  const int lane = threadIdx.x & 31, wave = threadIdx.x >> 5;
  const size_t row0 = ((size_t)blockIdx.x * 4 + wave) * 32;  // token rows = B*14
  float* s = &S1[wave][0];

  v16h ax[2][4];
#pragma unroll
  for (int m = 0; m < 2; ++m)
#pragma unroll
    for (int kb = 0; kb < 4; ++kb)
      ax[m][kb] = load_a_f32(x14 + (row0 + m * 16) * 128, 128, kb, lane);

  const int rb = (lane >> 4) * 8, cl = lane & 15;
#pragma unroll
  for (int nt = 0; nt < 8; ++nt) {
    v8f a0 = zero_c(), a1 = zero_c();
#pragma unroll
    for (int kb = 0; kb < 4; ++kb) {
      v16h b = load_b(fa_s, kb, nt, 8, lane);
      a0 = wmma_ff(ax[0][kb], b, a0);
      a1 = wmma_ff(ax[1][kb], b, a1);
    }
    const float bv = fa_b[nt * 16 + cl];
#pragma unroll
    for (int r = 0; r < 8; ++r) {
      s[(rb + r) * 128 + nt * 16 + cl]      = a0[r] + bv;
      s[(16 + rb + r) * 128 + nt * 16 + cl] = a1[r] + bv;
    }
  }
  __syncthreads();

  {  // one lane per row (all 32 lanes): softmax over 128 + multiply by x14
    const int row = lane;
    float rv[128];
#pragma unroll
    for (int c4 = 0; c4 < 32; ++c4) {
      v4f d = *(const v4f*)(s + row * 128 + c4 * 4);
#pragma unroll
      for (int i = 0; i < 4; ++i) rv[c4 * 4 + i] = d[i];
    }
    float mx = -3.0e38f;
#pragma unroll
    for (int c = 0; c < 128; ++c) mx = fmaxf(mx, rv[c]);
    float sm = 0.f;
#pragma unroll
    for (int c = 0; c < 128; ++c) { rv[c] = __expf(rv[c] - mx); sm += rv[c]; }
    const float inv = 1.f / sm;
    const float* xr = x14 + (row0 + row) * 128;
    h16* wr = seq + (row0 + row) * 128;
#pragma unroll
    for (int c8 = 0; c8 < 16; ++c8) {
      v4f x0 = *(const v4f*)(xr + c8 * 8);
      v4f x1 = *(const v4f*)(xr + c8 * 8 + 4);
      v8h o;
#pragma unroll
      for (int i = 0; i < 4; ++i) {
        o[i]     = (h16)(rv[c8 * 8 + i] * inv * x0[i]);
        o[4 + i] = (h16)(rv[c8 * 8 + 4 + i] * inv * x1[i]);
      }
      *(v8h*)(wr + c8 * 8) = o;
    }
  }
}

// ---------------------------------------------------------------------------
// Kernel 2: encoder LSTM (14 steps) + daily heads. seq: xw -> hs in place.
// 8 waves/WG, 32 samples per wave. MAC phase: static_for over 128 (j,gate,kb)
// tiles, depth-2 pipelined wih/whh B-tile loads, guaranteed straight-line.
// ---------------------------------------------------------------------------
__global__ void __launch_bounds__(256, 1)
k_enc(h16* __restrict__ seq, const h16* __restrict__ ewih_s,
      const h16* __restrict__ ewhh_s, const float* __restrict__ ebsum,
      const float* __restrict__ mu_w, const float* __restrict__ mu_b,
      const float* __restrict__ sig_w, const float* __restrict__ sig_b,
      float* __restrict__ mu_d, float* __restrict__ sig_d) {
  __shared__ h16 S2[8][32 * 128];
  const int lane = threadIdx.x & 31, wave = threadIdx.x >> 5;
  const size_t sample0 = ((size_t)blockIdx.x * 8 + wave) * 32;
  h16* s = &S2[wave][0];
  const int rb = (lane >> 4) * 8, cl = lane & 15;

  v16h ah[2][4];
  v8f cst[2][8];
#pragma unroll
  for (int m = 0; m < 2; ++m) {
#pragma unroll
    for (int kb = 0; kb < 4; ++kb) ah[m][kb] = zero_a();
#pragma unroll
    for (int j = 0; j < 8; ++j) cst[m][j] = zero_c();
  }

#pragma unroll 1
  for (int t = 0; t < T_; ++t) {
    v16h ax[2][4];
#pragma unroll
    for (int m = 0; m < 2; ++m)
#pragma unroll
      for (int kb = 0; kb < 4; ++kb)
        ax[m][kb] = load_a_f16(seq + ((sample0 + m * 16) * T_ + t) * 128,
                               T_ * 128, kb, lane);

    // Flattened MAC phase: v = j*16 + gt*4 + kb; depth-2 B pipeline.
    v16h bwB[2], bhB[2];
    bwB[0] = load_b(ewih_s, 0, 0, 32, lane);
    bhB[0] = load_b(ewhh_s, 0, 0, 32, lane);
    bwB[1] = load_b(ewih_s, 1, 0, 32, lane);
    bhB[1] = load_b(ewhh_s, 1, 0, 32, lane);
    v8f g[2][4];
    sfor<128>([&](auto vc) {
      constexpr int v = decltype(vc)::value;
      constexpr int j = v >> 4, u = v & 15, gt = u >> 2, kb = u & 3, p = v & 1;
      if constexpr (u == 0) {
#pragma unroll
        for (int m = 0; m < 2; ++m)
#pragma unroll
          for (int q = 0; q < 4; ++q) g[m][q] = zero_c();
      }
      v16h bw = bwB[p], bh = bhB[p];
      if constexpr (v + 2 < 128) {
        constexpr int v2 = v + 2, j2 = v2 >> 4, u2 = v2 & 15;
        constexpr int gt2 = u2 >> 2, kb2 = u2 & 3;
        bwB[p] = load_b(ewih_s, kb2, gt2 * 8 + j2, 32, lane);
        bhB[p] = load_b(ewhh_s, kb2, gt2 * 8 + j2, 32, lane);
      }
      g[0][gt] = wmma_ff(ax[0][kb], bw, g[0][gt]);
      g[1][gt] = wmma_ff(ax[1][kb], bw, g[1][gt]);
      g[0][gt] = wmma_ff(ah[0][kb], bh, g[0][gt]);
      g[1][gt] = wmma_ff(ah[1][kb], bh, g[1][gt]);
      if constexpr (u == 15) {  // gate epilogue for hidden tile j
        const float b0 = ebsum[0 * 128 + j * 16 + cl];
        const float b1 = ebsum[1 * 128 + j * 16 + cl];
        const float b2 = ebsum[2 * 128 + j * 16 + cl];
        const float b3 = ebsum[3 * 128 + j * 16 + cl];
#pragma unroll
        for (int m = 0; m < 2; ++m)
#pragma unroll
          for (int r = 0; r < 8; ++r) {
            float ii = sigm_f(g[m][0][r] + b0), ff = sigm_f(g[m][1][r] + b1);
            float gg = tanh_f(g[m][2][r] + b2), oo = sigm_f(g[m][3][r] + b3);
            float cn = ff * cst[m][j][r] + ii * gg;
            cst[m][j][r] = cn;
            s[(m * 16 + rb + r) * 128 + j * 16 + cl] = (h16)(oo * tanh_f(cn));
          }
      }
    });
    __syncthreads();
#pragma unroll
    for (int m = 0; m < 2; ++m)
#pragma unroll
      for (int kb = 0; kb < 4; ++kb)
        ah[m][kb] = load_a_f16(s + m * 16 * 128, 128, kb, lane);
    // coalesced hs writeback (overwrites xw row for this (b,t))
#pragma unroll
    for (int r2 = 0; r2 < 16; ++r2) {
      int row = r2 * 2 + (lane >> 4), ch = lane & 15;
      v8h d = *(v8h*)(s + row * 128 + ch * 8);
      *(v8h*)(seq + ((sample0 + row) * T_ + t) * 128 + ch * 8) = d;
    }
    __syncthreads();
  }
  // daily heads from h13 (still in LDS); one lane per row, all 32 lanes
  {
    const size_t b = sample0 + lane;
    float m = mu_b[0], g = sig_b[0];
    head_dot(s + lane * 128, mu_w, sig_w, m, g);
    mu_d[b] = m; sig_d[b] = softp_f(g);
  }
}

// ---------------------------------------------------------------------------
// Kernel 3: attention pooling over steps 7..13 -> monthly token (week1 is
// dead code in the reference). 4 waves/WG, 32 samples per wave; scalar
// phases run one sample per lane with v8h LDS accesses.
// ---------------------------------------------------------------------------
__global__ void __launch_bounds__(128, 1)
k_pool(const h16* __restrict__ seq, const h16* __restrict__ wk_s,
       const h16* __restrict__ wv_s, const h16* __restrict__ wout_s,
       const float* __restrict__ q_s, const float* __restrict__ pool_bin,
       const float* __restrict__ pool_bout, h16* __restrict__ tok) {
  __shared__ h16  KV[4][32 * 128];
  __shared__ float SC[4][32 * 32];      // [sample][head*8 + step]
  __shared__ float QS[128];
  const int lane = threadIdx.x & 31, wave = threadIdx.x >> 5;
  const size_t sample0 = ((size_t)blockIdx.x * 4 + wave) * 32;
  h16* s = &KV[wave][0];
  float* sc = &SC[wave][0];
  if (threadIdx.x < 128) QS[threadIdx.x] = q_s[threadIdx.x];
  __syncthreads();
  const int rb = (lane >> 4) * 8, cl = lane & 15;

  // ---- K phase: scores[b,h,st] = q_h . k / sqrt(32)
#pragma unroll 1
  for (int st = 0; st < 7; ++st) {
    v16h ax[2][4];
#pragma unroll
    for (int m = 0; m < 2; ++m)
#pragma unroll
      for (int kb = 0; kb < 4; ++kb)
        ax[m][kb] = load_a_f16(seq + ((sample0 + m * 16) * T_ + 7 + st) * 128,
                               T_ * 128, kb, lane);
#pragma unroll
    for (int nt = 0; nt < 8; ++nt) {
      v8f a0 = zero_c(), a1 = zero_c();
#pragma unroll
      for (int kb = 0; kb < 4; ++kb) {
        v16h b = load_b(wk_s, kb, nt, 8, lane);
        a0 = wmma_ff(ax[0][kb], b, a0);
        a1 = wmma_ff(ax[1][kb], b, a1);
      }
      const float bk = pool_bin[128 + nt * 16 + cl];
#pragma unroll
      for (int r = 0; r < 8; ++r) {
        s[(rb + r) * 128 + nt * 16 + cl]      = (h16)(a0[r] + bk);
        s[(16 + rb + r) * 128 + nt * 16 + cl] = (h16)(a1[r] + bk);
      }
    }
    __syncthreads();
    {
#pragma unroll
      for (int h = 0; h < 4; ++h) {
        float a = 0.f;
#pragma unroll
        for (int d8 = 0; d8 < 4; ++d8) {
          v8h hh = *(const v8h*)(s + lane * 128 + h * 32 + d8 * 8);
#pragma unroll
          for (int i = 0; i < 8; ++i)
            a += QS[h * 32 + d8 * 8 + i] * (float)hh[i];
        }
        sc[lane * 32 + h * 8 + st] = a * 0.17677669529663687f;  // 1/sqrt(32)
      }
    }
    __syncthreads();
  }
  // ---- softmax over 7 steps per (sample, head); one sample per lane
  {
#pragma unroll
    for (int h = 0; h < 4; ++h) {
      float mx = -3.0e38f;
      for (int st = 0; st < 7; ++st) mx = fmaxf(mx, sc[lane * 32 + h * 8 + st]);
      float sm = 0.f;
      for (int st = 0; st < 7; ++st) {
        float e = __expf(sc[lane * 32 + h * 8 + st] - mx);
        sc[lane * 32 + h * 8 + st] = e; sm += e;
      }
      float inv = 1.f / sm;
      for (int st = 0; st < 7; ++st) sc[lane * 32 + h * 8 + st] *= inv;
    }
  }
  __syncthreads();

  // ---- V phase: ctx accumulated in registers, one sample per lane
  float ctx[128];
#pragma unroll
  for (int k = 0; k < 128; ++k) ctx[k] = 0.f;
#pragma unroll 1
  for (int st = 0; st < 7; ++st) {
    v16h ax[2][4];
#pragma unroll
    for (int m = 0; m < 2; ++m)
#pragma unroll
      for (int kb = 0; kb < 4; ++kb)
        ax[m][kb] = load_a_f16(seq + ((sample0 + m * 16) * T_ + 7 + st) * 128,
                               T_ * 128, kb, lane);
#pragma unroll
    for (int nt = 0; nt < 8; ++nt) {
      v8f a0 = zero_c(), a1 = zero_c();
#pragma unroll
      for (int kb = 0; kb < 4; ++kb) {
        v16h b = load_b(wv_s, kb, nt, 8, lane);
        a0 = wmma_ff(ax[0][kb], b, a0);
        a1 = wmma_ff(ax[1][kb], b, a1);
      }
      const float bv = pool_bin[256 + nt * 16 + cl];
#pragma unroll
      for (int r = 0; r < 8; ++r) {
        s[(rb + r) * 128 + nt * 16 + cl]      = (h16)(a0[r] + bv);
        s[(16 + rb + r) * 128 + nt * 16 + cl] = (h16)(a1[r] + bv);
      }
    }
    __syncthreads();
    {
#pragma unroll
      for (int h = 0; h < 4; ++h) {
        float w = sc[lane * 32 + h * 8 + st];
#pragma unroll
        for (int d8 = 0; d8 < 4; ++d8) {
          v8h hh = *(const v8h*)(s + lane * 128 + h * 32 + d8 * 8);
#pragma unroll
          for (int i = 0; i < 8; ++i)
            ctx[h * 32 + d8 * 8 + i] += w * (float)hh[i];
        }
      }
    }
    __syncthreads();
  }
  // ---- out projection: token = ctx @ wout.T + bout
  {
#pragma unroll
    for (int c8 = 0; c8 < 16; ++c8) {
      v8h o;
#pragma unroll
      for (int i = 0; i < 8; ++i) o[i] = (h16)ctx[c8 * 8 + i];
      *(v8h*)(s + lane * 128 + c8 * 8) = o;
    }
  }
  __syncthreads();
  v16h axc[2][4];
#pragma unroll
  for (int m = 0; m < 2; ++m)
#pragma unroll
    for (int kb = 0; kb < 4; ++kb)
      axc[m][kb] = load_a_f16(s + m * 16 * 128, 128, kb, lane);
#pragma unroll
  for (int nt = 0; nt < 8; ++nt) {
    v8f a0 = zero_c(), a1 = zero_c();
#pragma unroll
    for (int kb = 0; kb < 4; ++kb) {
      v16h b = load_b(wout_s, kb, nt, 8, lane);
      a0 = wmma_ff(axc[0][kb], b, a0);
      a1 = wmma_ff(axc[1][kb], b, a1);
    }
    const float bo = pool_bout[nt * 16 + cl];
#pragma unroll
    for (int r = 0; r < 8; ++r) {
      tok[(sample0 + rb + r) * 128 + nt * 16 + cl]      = (h16)(a0[r] + bo);
      tok[(sample0 + 16 + rb + r) * 128 + nt * 16 + cl] = (h16)(a1[r] + bo);
    }
  }
}

// ---------------------------------------------------------------------------
// Kernel 4: decoder LSTM (30 steps) + per-step heads.
// Step 0: g = tok @ whh.T + b;   steps>=1 (x_t == h_t): g = h @ (wih+whh).T + b
// 8 waves/WG, 32 samples per wave; static_for MAC loop, depth-2 B pipeline.
// ---------------------------------------------------------------------------
__global__ void __launch_bounds__(256, 1)
k_dec(const h16* __restrict__ tok, const h16* __restrict__ dwhh_s,
      const h16* __restrict__ dwsum_s, const float* __restrict__ dbsum,
      const float* __restrict__ dmu_w, const float* __restrict__ dmu_b,
      const float* __restrict__ dsig_w, const float* __restrict__ dsig_b,
      float* __restrict__ mu_m, float* __restrict__ sig_m) {
  __shared__ h16 S4[8][32 * 128];
  const int lane = threadIdx.x & 31, wave = threadIdx.x >> 5;
  const size_t sample0 = ((size_t)blockIdx.x * 8 + wave) * 32;
  h16* s = &S4[wave][0];
  const int rb = (lane >> 4) * 8, cl = lane & 15;

  v16h ah[2][4];
  v8f cst[2][8];
#pragma unroll
  for (int m = 0; m < 2; ++m) {
#pragma unroll
    for (int kb = 0; kb < 4; ++kb)
      ah[m][kb] = load_a_f16(tok + (sample0 + m * 16) * 128, 128, kb, lane);
#pragma unroll
    for (int j = 0; j < 8; ++j) cst[m][j] = zero_c();
  }

#pragma unroll 1
  for (int t = 0; t < ST_; ++t) {
    const h16* bsel = (t == 0) ? dwhh_s : dwsum_s;
    v16h bB[2];
    bB[0] = load_b(bsel, 0, 0, 32, lane);
    bB[1] = load_b(bsel, 1, 0, 32, lane);
    v8f g[2][4];
    sfor<128>([&](auto vc) {
      constexpr int v = decltype(vc)::value;
      constexpr int j = v >> 4, u = v & 15, gt = u >> 2, kb = u & 3, p = v & 1;
      if constexpr (u == 0) {
#pragma unroll
        for (int m = 0; m < 2; ++m)
#pragma unroll
          for (int q = 0; q < 4; ++q) g[m][q] = zero_c();
      }
      v16h b = bB[p];
      if constexpr (v + 2 < 128) {
        constexpr int v2 = v + 2, j2 = v2 >> 4, u2 = v2 & 15;
        constexpr int gt2 = u2 >> 2, kb2 = u2 & 3;
        bB[p] = load_b(bsel, kb2, gt2 * 8 + j2, 32, lane);
      }
      g[0][gt] = wmma_ff(ah[0][kb], b, g[0][gt]);
      g[1][gt] = wmma_ff(ah[1][kb], b, g[1][gt]);
      if constexpr (u == 15) {  // gate epilogue for hidden tile j
        const float b0 = dbsum[0 * 128 + j * 16 + cl];
        const float b1 = dbsum[1 * 128 + j * 16 + cl];
        const float b2 = dbsum[2 * 128 + j * 16 + cl];
        const float b3 = dbsum[3 * 128 + j * 16 + cl];
#pragma unroll
        for (int m = 0; m < 2; ++m)
#pragma unroll
          for (int r = 0; r < 8; ++r) {
            float ii = sigm_f(g[m][0][r] + b0), ff = sigm_f(g[m][1][r] + b1);
            float gg = tanh_f(g[m][2][r] + b2), oo = sigm_f(g[m][3][r] + b3);
            float cn = ff * cst[m][j][r] + ii * gg;
            cst[m][j][r] = cn;
            s[(m * 16 + rb + r) * 128 + j * 16 + cl] = (h16)(oo * tanh_f(cn));
          }
      }
    });
    __syncthreads();
#pragma unroll
    for (int m = 0; m < 2; ++m)
#pragma unroll
      for (int kb = 0; kb < 4; ++kb)
        ah[m][kb] = load_a_f16(s + m * 16 * 128, 128, kb, lane);
    {  // per-step heads: one sample per lane, all 32 lanes
      const size_t b = sample0 + lane;
      float m = dmu_b[0], g = dsig_b[0];
      head_dot(s + lane * 128, dmu_w, dsig_w, m, g);
      mu_m[b * ST_ + t] = m; sig_m[b * ST_ + t] = softp_f(g);
    }
    __syncthreads();
  }
}

// ---------------------------------------------------------------------------
extern "C" void kernel_launch(void* const* d_in, const int* in_sizes, int n_in,
                              void* d_out, int out_size, void* d_ws, size_t ws_size,
                              hipStream_t stream) {
  const float* x14      = (const float*)d_in[0];
  const float* fa_w     = (const float*)d_in[1];
  const float* fa_b     = (const float*)d_in[2];
  const float* enc_wih  = (const float*)d_in[3];
  const float* enc_whh  = (const float*)d_in[4];
  const float* enc_bih  = (const float*)d_in[5];
  const float* enc_bhh  = (const float*)d_in[6];
  const float* q_tok    = (const float*)d_in[7];
  const float* pool_win = (const float*)d_in[8];
  const float* pool_bin = (const float*)d_in[9];
  const float* pool_wout= (const float*)d_in[10];
  const float* pool_bout= (const float*)d_in[11];
  const float* mu_w     = (const float*)d_in[12];
  const float* mu_b     = (const float*)d_in[13];
  const float* sig_w    = (const float*)d_in[14];
  const float* sig_b    = (const float*)d_in[15];
  const float* dec_wih  = (const float*)d_in[16];
  const float* dec_whh  = (const float*)d_in[17];
  const float* dec_bih  = (const float*)d_in[18];
  const float* dec_bhh  = (const float*)d_in[19];
  const float* dmu_w    = (const float*)d_in[20];
  const float* dmu_b    = (const float*)d_in[21];
  const float* dsig_w   = (const float*)d_in[22];
  const float* dsig_b   = (const float*)d_in[23];

  char* ws = (char*)d_ws;
  h16*   seq    = (h16*)(ws + OFF_SEQ);
  h16*   tok    = (h16*)(ws + OFF_TOK);
  h16*   fa_s   = (h16*)(ws + OFF_FA);
  h16*   ewih_s = (h16*)(ws + OFF_EWIH);
  h16*   ewhh_s = (h16*)(ws + OFF_EWHH);
  h16*   wk_s   = (h16*)(ws + OFF_WK);
  h16*   wv_s   = (h16*)(ws + OFF_WV);
  h16*   wout_s = (h16*)(ws + OFF_WOUT);
  h16*   dwhh_s = (h16*)(ws + OFF_DWHH);
  h16*   dwsum_s= (h16*)(ws + OFF_DWSUM);
  float* q_s    = (float*)(ws + OFF_Q);
  float* ebsum_s= (float*)(ws + OFF_EBSUM);
  float* dbsum_s= (float*)(ws + OFF_DBSUM);

  float* out   = (float*)d_out;
  float* mu_d  = out;
  float* sig_d = out + B_;
  float* mu_m  = out + 2 * B_;
  float* sig_m = out + 2 * B_ + (size_t)B_ * ST_;

  k_prep<<<96, 256, 0, stream>>>(fa_w, enc_wih, enc_whh, pool_win, pool_wout,
                                 dec_wih, dec_whh, q_tok, pool_bin,
                                 enc_bih, enc_bhh, dec_bih, dec_bhh,
                                 fa_s, ewih_s, ewhh_s, wk_s, wv_s, wout_s,
                                 dwhh_s, dwsum_s, q_s, ebsum_s, dbsum_s);
  k_feat<<<(B_ * T_) / 128, 128, 0, stream>>>(x14, fa_s, fa_b, seq);
  k_enc<<<B_ / 256, 256, 0, stream>>>(seq, ewih_s, ewhh_s, ebsum_s,
                                      mu_w, mu_b, sig_w, sig_b, mu_d, sig_d);
  k_pool<<<B_ / 128, 128, 0, stream>>>(seq, wk_s, wv_s, wout_s, q_s,
                                       pool_bin, pool_bout, tok);
  k_dec<<<B_ / 256, 256, 0, stream>>>(tok, dwhh_s, dwsum_s, dbsum_s,
                                      dmu_w, dmu_b, dsig_w, dsig_b, mu_m, sig_m);
}
